// MultiHeadAttnBlock_3968549781788
// MI455X (gfx1250) — matrix operand order repaired and checked by reading, not compiled, not run on hardware
//
#include <hip/hip_runtime.h>

// MI455X / gfx1250 multi-head attention block.
// f16 WMMA (v_wmma_f32_16x16x32_f16) for all GEMMs, flash-attention streaming
// softmax so the 4x4096x4096 score tensor never touches HBM. K/V tiles are
// streamed into LDS with gfx1250 async global->LDS loads (ASYNCcnt) and
// double-buffered so tile t+1 loads overlap tile t's WMMAs.

typedef __attribute__((ext_vector_type(16))) _Float16 v16h;
typedef __attribute__((ext_vector_type(8)))  _Float16 v8h;
typedef __attribute__((ext_vector_type(8)))  float    v8f;

#define C_DIM 256
#define N_TOK 4096
#define HD    64
#define NH    4

static __device__ inline v16h cat16(v8h lo, v8h hi) {
  return __builtin_shufflevector(lo, hi, 0,1,2,3,4,5,6,7,8,9,10,11,12,13,14,15);
}

static __device__ inline v8f wmma_f16(v16h a, v16h b, v8f c) {
  // (neg_a, A, neg_b, B, c_mod, C, reuse_a, reuse_b)
  return __builtin_amdgcn_wmma_f32_16x16x32_f16(false, a, false, b, (short)0, c,
                                                false, false);
}

// A-matrix 16x32 f16 fragment: lane<16 holds K={kb..kb+7, kb+16..kb+23},
// lane>=16 holds K={kb+8..kb+15, kb+24..kb+31}; row M = lane&15 (row ptr given).
static __device__ inline v16h load_afrag(const _Float16* row, int kb, int lane) {
  int c0 = kb + ((lane & 16) ? 8 : 0);
  int c1 = kb + ((lane & 16) ? 24 : 16);
  v8h lo = *(const v8h*)(row + c0);
  v8h hi = *(const v8h*)(row + c1);
  return cat16(lo, hi);
}

// B-matrix 32x16 f16 fragment: lane holds column N = lane&15 (col ptr given is
// that column laid out contiguous in K); lane<16 K={kb..kb+7, kb+16..kb+23},
// lane>=16 K={kb+8..kb+15, kb+24..kb+31}.
static __device__ inline v16h load_bfrag(const _Float16* col, int kb, int lane) {
  int c0 = kb + ((lane & 16) ? 8 : 0);
  int c1 = kb + ((lane & 16) ? 24 : 16);
  v8h lo = *(const v8h*)(col + c0);
  v8h hi = *(const v8h*)(col + c1);
  return cat16(lo, hi);
}

// gfx1250 async global->LDS copy, 16B per lane, tracked by ASYNCcnt.
// lds_off = LDS byte offset (low 32 bits of a generic pointer to __shared__).
static __device__ inline void async_copy_b128(unsigned lds_off, const void* gaddr) {
  asm volatile("global_load_async_to_lds_b128 %0, %1, off"
               :: "v"(lds_off), "v"(gaddr) : "memory");
}
static __device__ inline void wait_async0() {
  asm volatile("s_wait_asynccnt 0" ::: "memory");
}

// ---------------------------------------------------------------- weights f32->f16
__global__ void wconv_kernel(const float* __restrict__ W0, const float* __restrict__ W1,
                             const float* __restrict__ W2, const float* __restrict__ W3,
                             _Float16* __restrict__ out) {
  int t = blockIdx.x * 256 + threadIdx.x;           // 4 * 65536 total
  int w = t >> 16, e = t & 65535;
  const float* src = (w == 0) ? W0 : (w == 1) ? W1 : (w == 2) ? W2 : W3;
  out[t] = (_Float16)src[e];
}

// ---------------------------------------------------------------- GroupNorm -> Ht[p][c] f16
__global__ void gn_kernel(const float* __restrict__ x, const float* __restrict__ gamma,
                          const float* __restrict__ beta, _Float16* __restrict__ Ht) {
  int g = blockIdx.x;                // 32 groups of 8 channels
  int tid = threadIdx.x;
  int c0 = g * 8;
  __shared__ float s_sum[256], s_sq[256];
  const float* xg = x + (size_t)c0 * N_TOK;
  float sum = 0.f, sq = 0.f;
  for (int e = tid; e < 8 * N_TOK; e += 256) {
    float v = xg[e];
    sum += v; sq += v * v;
  }
  s_sum[tid] = sum; s_sq[tid] = sq;
  __syncthreads();
  for (int s = 128; s > 0; s >>= 1) {
    if (tid < s) { s_sum[tid] += s_sum[tid + s]; s_sq[tid] += s_sq[tid + s]; }
    __syncthreads();
  }
  float mean = s_sum[0] * (1.0f / 32768.0f);
  float var  = s_sq[0] * (1.0f / 32768.0f) - mean * mean;
  float rstd = rsqrtf(var + 1e-6f);
  for (int e = tid; e < 8 * N_TOK; e += 256) {
    int cl = e >> 12, p = e & 4095;
    int c = c0 + cl;
    float v = (xg[e] - mean) * rstd * gamma[c] + beta[c];
    Ht[(size_t)p * C_DIM + c] = (_Float16)v;   // token-major for GEMM B-frags
  }
}

// ---------------------------------------------------------------- QKV projection GEMM
// out[o,p] = sum_c W[o,c] * Ht[p,c] (+bias). One wave per 16x16 tile, K=256 in 8 steps.
__global__ void __launch_bounds__(128)
qkv_kernel(const _Float16* __restrict__ Wh, const float* __restrict__ bq,
           const float* __restrict__ bk, const float* __restrict__ bv,
           const _Float16* __restrict__ Ht, _Float16* __restrict__ Qt,
           _Float16* __restrict__ Kt, _Float16* __restrict__ Vd) {
  int which = blockIdx.z;                 // 0=q 1=k 2=v
  int lane = threadIdx.x & 31, wv = threadIdx.x >> 5;
  int lu = lane & 15, half8 = (lane & 16) ? 8 : 0;
  int mt = blockIdx.y * 4 + wv;           // 16 M-tiles
  int nt = blockIdx.x;                    // 256 N-tiles
  const _Float16* W = Wh + (size_t)which * 65536;
  const _Float16* arow = W  + (size_t)(mt * 16 + lu) * C_DIM;
  const _Float16* brow = Ht + (size_t)(nt * 16 + lu) * C_DIM;
  v8f acc = {};
#pragma unroll
  for (int ks = 0; ks < 8; ++ks)
    acc = wmma_f16(load_afrag(arow, ks * 32, lane), load_bfrag(brow, ks * 32, lane), acc);

  int p  = nt * 16 + lu;
  int o0 = mt * 16 + half8;               // rows o0..o0+7 for this lane
  const float* bias = (which == 0) ? bq : (which == 1) ? bk : bv;
  if (which < 2) {
    // token-major [h][n][hd]; Q pre-scaled by hd^-0.5
    float qscale = (which == 0) ? 0.125f : 1.0f;
    int h = mt >> 2;
    int d0 = o0 & 63;                     // contiguous d for r=0..7
    v8h outv;
#pragma unroll
    for (int r = 0; r < 8; ++r)
      outv[r] = (_Float16)((acc[r] + bias[o0 + r]) * qscale);
    _Float16* dst = ((which == 0) ? Qt : Kt) + ((size_t)(h * N_TOK + p) * HD + d0);
    *(v8h*)dst = outv;
  } else {
    // V channel-major [h][hd][n] == [o][p]
#pragma unroll
    for (int r = 0; r < 8; ++r)
      Vd[(size_t)(o0 + r) * N_TOK + p] = (_Float16)(acc[r] + bias[o0 + r]);
  }
}

// ---------------------------------------------------------------- flash attention
// Block = 4 waves, each wave owns a 16-row stripe of a 64-row tile of one head.
// K/V 64-column tiles double-buffered in LDS via async global->LDS loads.
static __device__ inline void issue_kv_tile(const _Float16* __restrict__ Kt,
                                            const _Float16* __restrict__ Vd,
                                            int h, int jb,
                                            _Float16* bufk, _Float16* bufv, int tid) {
#pragma unroll
  for (int c0 = 0; c0 < 512; c0 += 128) {
    int c = c0 + tid;
    int row = c >> 3, off = (c & 7) * 8;
    async_copy_b128((unsigned)(size_t)(bufk + row * 64 + off),
                    Kt + (size_t)(h * N_TOK + jb + row) * HD + off);
    async_copy_b128((unsigned)(size_t)(bufv + row * 64 + off),
                    Vd + (size_t)(h * HD + row) * N_TOK + jb + off);
  }
}

__global__ void __launch_bounds__(128)
attn_kernel(const _Float16* __restrict__ Qt, const _Float16* __restrict__ Kt,
            const _Float16* __restrict__ Vd, _Float16* __restrict__ OB) {
  __shared__ __align__(16) _Float16 lds_k[2][64 * 64];   // Kt rows jb..jb+63
  __shared__ __align__(16) _Float16 lds_v[2][64 * 64];   // Vd[d][jb..jb+63]
  __shared__ __align__(16) _Float16 lds_p[4][16 * 64];   // per-wave P staging

  int h = blockIdx.y;
  int rowbase = blockIdx.x * 64;
  int tid = threadIdx.x;
  int lane = tid & 31, wv = tid >> 5;
  int lu = lane & 15, half8 = (lane & 16) ? 8 : 0;

  // Q A-fragments for this wave's rows (K-dim = hd = 64 -> 2 ksteps)
  const _Float16* qrow = Qt + (size_t)(h * N_TOK + rowbase + wv * 16 + lu) * HD;
  v16h qa0 = load_afrag(qrow, 0, lane);
  v16h qa1 = load_afrag(qrow, 32, lane);

  v8f acc[4] = {};                 // 16 rows x 64 d (4 N-tiles)
  float m[8], l[8];
#pragma unroll
  for (int r = 0; r < 8; ++r) { m[r] = -1e30f; l[r] = 0.f; }

  issue_kv_tile(Kt, Vd, h, 0, lds_k[0], lds_v[0], tid);   // prologue: tile 0

  for (int it = 0; it < N_TOK / 64; ++it) {
    int cur = it & 1;
    wait_async0();        // my async loads for tile 'it' landed in LDS
    __syncthreads();      // all waves' loads landed; buf cur^1 fully consumed
    if (it + 1 < N_TOK / 64)
      issue_kv_tile(Kt, Vd, h, (it + 1) * 64, lds_k[cur ^ 1], lds_v[cur ^ 1], tid);
    const _Float16* lk = lds_k[cur];
    const _Float16* lv = lds_v[cur];

    // S = Q K^T : 4 j-tiles of 16 cols, K-dim = 64
    v8f s[4];
#pragma unroll
    for (int jt = 0; jt < 4; ++jt) {
      const _Float16* kcol = lk + (jt * 16 + lu) * 64;
      v8f z = {};
      z = wmma_f16(qa0, load_bfrag(kcol, 0, lane), z);
      s[jt] = wmma_f16(qa1, load_bfrag(kcol, 32, lane), z);
    }

    // online softmax; lanes 0-15 share rows 0-7, lanes 16-31 share rows 8-15
    _Float16* pl = lds_p[wv];
#pragma unroll
    for (int r = 0; r < 8; ++r) {
      float mx = fmaxf(fmaxf(s[0][r], s[1][r]), fmaxf(s[2][r], s[3][r]));
#pragma unroll
      for (int d = 8; d >= 1; d >>= 1) mx = fmaxf(mx, __shfl_xor(mx, d, 32));
      float mnew = fmaxf(m[r], mx);
      float corr = __expf(m[r] - mnew);
      float rsum = 0.f;
#pragma unroll
      for (int jt = 0; jt < 4; ++jt) {
        float pv = __expf(s[jt][r] - mnew);
        rsum += pv;
        pl[(r + half8) * 64 + jt * 16 + lu] = (_Float16)pv;   // row-major P
      }
#pragma unroll
      for (int d = 8; d >= 1; d >>= 1) rsum += __shfl_xor(rsum, d, 32);
      l[r] = l[r] * corr + rsum;
      m[r] = mnew;
#pragma unroll
      for (int dt = 0; dt < 4; ++dt) acc[dt][r] *= corr;
    }

    asm volatile("s_wait_dscnt 0" ::: "memory");  // wave-local P store->load

    // P A-frags (K-dim = 64 cols) then acc += P * V^T
    const _Float16* prow = pl + lu * 64;
    v16h pa0 = load_afrag(prow, 0, lane);
    v16h pa1 = load_afrag(prow, 32, lane);
#pragma unroll
    for (int dt = 0; dt < 4; ++dt) {
      const _Float16* vcol = lv + (dt * 16 + lu) * 64;
      acc[dt] = wmma_f16(pa0, load_bfrag(vcol, 0, lane), acc[dt]);
      acc[dt] = wmma_f16(pa1, load_bfrag(vcol, 32, lane), acc[dt]);
    }
  }

  // Torch-faithful reshape: flat i*256 + c' == c*4096 + p. Store transposed as
  // OB[p][c] (c contiguous) so projection B-frags are contiguous:
  //   c = i>>4 ; p = (i&15)*256 + c'
#pragma unroll
  for (int r = 0; r < 8; ++r) {
    int i = rowbase + wv * 16 + r + half8;
    float inv = 1.0f / l[r];
#pragma unroll
    for (int dt = 0; dt < 4; ++dt) {
      int cp = h * 64 + dt * 16 + lu;
      OB[(size_t)((i & 15) * 256 + cp) * 256 + (i >> 4)] = (_Float16)(acc[dt][r] * inv);
    }
  }
}

// ---------------------------------------------------------------- projection + residual
__global__ void __launch_bounds__(128)
proj_kernel(const _Float16* __restrict__ Wp, const float* __restrict__ bp,
            const _Float16* __restrict__ OB, const float* __restrict__ x,
            float* __restrict__ y) {
  int lane = threadIdx.x & 31, wv = threadIdx.x >> 5;
  int lu = lane & 15, half8 = (lane & 16) ? 8 : 0;
  int mt = blockIdx.y * 4 + wv, nt = blockIdx.x;
  const _Float16* arow = Wp + (size_t)(mt * 16 + lu) * C_DIM;
  const _Float16* brow = OB + (size_t)(nt * 16 + lu) * C_DIM;
  v8f acc = {};
#pragma unroll
  for (int ks = 0; ks < 8; ++ks)
    acc = wmma_f16(load_afrag(arow, ks * 32, lane), load_bfrag(brow, ks * 32, lane), acc);
  int p = nt * 16 + lu;
  int o0 = mt * 16 + half8;
#pragma unroll
  for (int r = 0; r < 8; ++r) {
    int o = o0 + r;
    y[(size_t)o * N_TOK + p] = acc[r] + bp[o] + x[(size_t)o * N_TOK + p];
  }
}

extern "C" void kernel_launch(void* const* d_in, const int* in_sizes, int n_in,
                              void* d_out, int out_size, void* d_ws, size_t ws_size,
                              hipStream_t stream) {
  (void)in_sizes; (void)n_in; (void)out_size; (void)ws_size;
  const float* x     = (const float*)d_in[0];
  const float* gamma = (const float*)d_in[1];
  const float* beta  = (const float*)d_in[2];
  const float* Wq    = (const float*)d_in[3];
  const float* bq    = (const float*)d_in[4];
  const float* Wk    = (const float*)d_in[5];
  const float* bk    = (const float*)d_in[6];
  const float* Wv    = (const float*)d_in[7];
  const float* bv    = (const float*)d_in[8];
  const float* Wp    = (const float*)d_in[9];
  const float* bp    = (const float*)d_in[10];
  float* y = (float*)d_out;

  _Float16* ws = (_Float16*)d_ws;          // ~10.5 MB of f16 scratch
  _Float16* Ht = ws;                       // [4096][256]
  _Float16* Wh = Ht + 4096 * 256;          // [4][256][256] q,k,v,p
  _Float16* Qt = Wh + 4 * 65536;           // [4][4096][64]
  _Float16* Kt = Qt + 4 * 4096 * 64;       // [4][4096][64]
  _Float16* Vd = Kt + 4 * 4096 * 64;       // [4][64][4096]
  _Float16* OB = Vd + 4 * 4096 * 64;       // [4096][256] transposed reshuffled attn out

  hipLaunchKernelGGL(wconv_kernel, dim3(1024), dim3(256), 0, stream, Wq, Wk, Wv, Wp, Wh);
  hipLaunchKernelGGL(gn_kernel, dim3(32), dim3(256), 0, stream, x, gamma, beta, Ht);
  hipLaunchKernelGGL(qkv_kernel, dim3(256, 4, 3), dim3(128), 0, stream,
                     Wh, bq, bk, bv, Ht, Qt, Kt, Vd);
  hipLaunchKernelGGL(attn_kernel, dim3(64, 4), dim3(128), 0, stream, Qt, Kt, Vd, OB);
  hipLaunchKernelGGL(proj_kernel, dim3(256, 4), dim3(128), 0, stream,
                     Wh + 3 * 65536, bp, OB, x, y);
}
